// GINConvNet_75024488726799
// MI455X (gfx1250) — matrix operand order repaired and checked by reading, not compiled, not run on hardware
//
#include <hip/hip_runtime.h>
#include <math.h>

#if defined(__AMDGCN__) && __has_builtin(__builtin_amdgcn_global_load_async_to_lds_b128)
#define HAS_ASYNC_LDS 1
#else
#define HAS_ASYNC_LDS 0
#endif

// ---------------------------------------------------------------------------
// CDNA5 WMMA helpers (bf16 in, f32 accumulate), wave32 layouts per ISA 7.12.2
// ---------------------------------------------------------------------------
typedef __attribute__((ext_vector_type(16))) __bf16 v16bf;
typedef __attribute__((ext_vector_type(8)))  __bf16 v8bf;
typedef __attribute__((ext_vector_type(8)))  float  v8f;
typedef __attribute__((ext_vector_type(4)))  int    v4i;

__device__ inline v8f wmma_bf16(v16bf a, v16bf b, v8f c) {
  return __builtin_amdgcn_wmma_f32_16x16x32_bf16(
      false, a, false, b, (short)0, c, false, false);
}

__device__ inline v16bf combine16(v8bf lo, v8bf hi) {
  return __builtin_shufflevector(lo, hi, 0, 1, 2, 3, 4, 5, 6, 7,
                                 8, 9, 10, 11, 12, 13, 14, 15);
}

// A fragment for one 16x32 tile; Ar = base of this lane's row (row = lane&15).
__device__ inline v16bf frag_a_rowptr(const __bf16* Ar, int lane, int k0) {
  int g = (lane >> 4) & 1;
  v8bf lo = *(const v8bf*)(Ar + k0 + (g << 3));
  v8bf hi = *(const v8bf*)(Ar + k0 + 16 + (g << 3));
  return combine16(lo, hi);
}

// B fragment from weights packed in fragment order: [kt][nt][lane][16 halves].
__device__ inline v16bf frag_b_packed(const __bf16* __restrict__ Wp, int lane,
                                      int kt, int nt, int ntilesN) {
  const __bf16* p = Wp + ((((kt * ntilesN + nt) << 5) + lane) << 4);
  v8bf lo = *(const v8bf*)p;
  v8bf hi = *(const v8bf*)(p + 8);
  return combine16(lo, hi);
}

__device__ inline void atomic_max_f32(float* addr, float v) {
  if (v >= 0.0f) atomicMax((int*)addr, __float_as_int(v));
  else           atomicMin((unsigned int*)addr, __float_as_uint(v));
}

#if HAS_ASYNC_LDS
__device__ inline void async_copy16(const __bf16* g, __bf16* l) {
  __builtin_amdgcn_global_load_async_to_lds_b128(
      (__attribute__((address_space(1))) v4i*)g,
      (__attribute__((address_space(3))) v4i*)l, 0, 0);
}
#endif

// ---------------------------------------------------------------------------
// Utility kernels
// ---------------------------------------------------------------------------
__global__ void k_fill(float* p, long n, float v) {
  long i = (long)blockIdx.x * blockDim.x + threadIdx.x;
  if (i < n) p[i] = v;
}

// pack fp32 (K x Nl) weight -> zero-padded bf16 (Kp x Np) in WMMA fragment order
__global__ void k_pack_wfrag(const float* __restrict__ src, int K, int Nl,
                             __bf16* __restrict__ dst, int Kp, int Np) {
  int i = blockIdx.x * blockDim.x + threadIdx.x;
  if (i >= Kp * Np) return;
  int e    = i & 15;
  int lane = (i >> 4) & 31;
  int tile = i >> 9;
  int ntN  = Np >> 4;
  int kt = tile / ntN, nt = tile - kt * ntN;
  int k = (kt << 5) + ((e >> 3) << 4) + ((lane >> 4) << 3) + (e & 7);
  int n = (nt << 4) + (lane & 15);
  float v = (k < K && n < Nl) ? src[k * Nl + n] : 0.0f;
  dst[i] = (__bf16)v;
}

// node features: grid(1, N), 96 threads
__global__ void k_nodefeat(const int* __restrict__ xn,
                           const float* __restrict__ e1, const float* __restrict__ e2,
                           const float* __restrict__ e3, const float* __restrict__ e4,
                           __bf16* __restrict__ h) {
  int n = blockIdx.y, c = threadIdx.x;
  float v = 0.0f;
  if (c < 32)       v = e1[xn[n * 8 + 0] * 32 + c];
  else if (c < 64) {
    int cc = c - 32;
    v = e2[xn[n * 8 + 1] * 32 + cc] + e3[xn[n * 8 + 2] * 32 + cc] + e4[xn[n * 8 + 3] * 32 + cc];
  } else if (c == 64) v = (float)(xn[n * 8 + 4] - 1);
  h[n * 96 + c] = (__bf16)v;
}

// edge features ea = [edge_emb(15) | type1-1 | zero-pad] -> E x 32 bf16
__global__ void k_edgefeat(const int* __restrict__ et, const float* __restrict__ emb,
                           __bf16* __restrict__ ea, int E) {
  int i = blockIdx.x * blockDim.x + threadIdx.x;
  if (i >= E * 32) return;
  int e = i >> 5, k = i & 31;
  float v = 0.0f;
  if (k < 15)       v = emb[et[e * 7 + 0] * 15 + k];
  else if (k == 15) v = (float)(et[e * 7 + 1] - 1);
  ea[i] = (__bf16)v;
}

// pos (N x 3 f32) -> N x 4 bf16
__global__ void k_posb(const float* __restrict__ pos, __bf16* __restrict__ pb, int N) {
  int i = blockIdx.x * blockDim.x + threadIdx.x;
  if (i >= N * 4) return;
  int n = i >> 2, c = i & 3;
  pb[i] = (__bf16)(c < 3 ? pos[n * 3 + c] : 0.0f);
}

// ---------------------------------------------------------------------------
// GINE edge messages via WMMA: per 16-edge block, m = ea(16x32) @ lw(32xKip);
// epilogue fuses + x[src] gather, + lb, relu, and atomicAdd segment-sum on dst.
// ---------------------------------------------------------------------------
__global__ __launch_bounds__(256) void k_gine_msg(
    const __bf16* __restrict__ X, int ldx,
    const __bf16* __restrict__ ea32,
    const __bf16* __restrict__ LWp, const float* __restrict__ lb,
    int ilen, int ntiles,
    const int* __restrict__ src, const int* __restrict__ dst, int E,
    float* __restrict__ agg, int ldagg) {
  __shared__ int sSrcOff[16], sDstOff[16];
  int tid = threadIdx.x;
  int e0 = blockIdx.x << 4;
  if (tid < 16) {
    int e = e0 + tid;
    if (e >= E) e = E - 1;
    sSrcOff[tid] = src[e] * ldx;
    sDstOff[tid] = dst[e] * ldagg;
  }
  __syncthreads();
  int wave = tid >> 5, lane = tid & 31;
  int arow = e0 + (lane & 15);
  if (arow >= E) arow = E - 1;
  v16bf fa = frag_a_rowptr(ea32 + arow * 32, lane, 0);
  bool full = (e0 + 16 <= E);
  int base = (lane >> 4) << 3;
  for (int nt = wave; nt < ntiles; nt += 8) {
    v16bf fb = frag_b_packed(LWp, lane, 0, nt, ntiles);
    v8f acc = (v8f){0, 0, 0, 0, 0, 0, 0, 0};
    acc = wmma_bf16(fa, fb, acc);
    int col = (nt << 4) + (lane & 15);
    if (col < ilen) {
      float bv = lb[col];
      if (full) {
#pragma unroll
        for (int r = 0; r < 8; ++r) {
          int row = base + r;
          float m = acc[r] + bv + (float)X[sSrcOff[row] + col];
          atomicAdd(&agg[sDstOff[row] + col], fmaxf(m, 0.0f));
        }
      } else {
#pragma unroll
        for (int r = 0; r < 8; ++r) {
          int row = base + r;
          if (e0 + row < E) {
            float m = acc[r] + bv + (float)X[sSrcOff[row] + col];
            atomicAdd(&agg[sDstOff[row] + col], fmaxf(m, 0.0f));
          }
        }
      }
    }
  }
}

// t = (1+eps)*x + agg; grid(1, N), Kp threads
__global__ void k_combine(const __bf16* __restrict__ X, int ldx,
                          const float* __restrict__ agg, int ldagg,
                          const float* __restrict__ epsp, int ilen, int Kp,
                          __bf16* __restrict__ T) {
  int n = blockIdx.y, c = threadIdx.x;
  float v = 0.0f;
  if (c < ilen) v = (1.0f + epsp[0]) * (float)X[n * ldx + c] + agg[n * ldagg + c];
  T[n * Kp + c] = (__bf16)v;
}

// ---------------------------------------------------------------------------
// Generic WMMA GEMM: one wave per 16x16 tile of C = act(A[MxK] @ Wp + bias)
// ---------------------------------------------------------------------------
__global__ __launch_bounds__(32) void k_gemm(
    const __bf16* __restrict__ A, int lda,
    const __bf16* __restrict__ Wp, int ntilesN,
    const float* __restrict__ bias, int Nlogic,
    int M, int Klen,
    __bf16* outB, int ldob, float* outF, int ldof, int relu) {
  int lane = threadIdx.x & 31;
  int m0 = blockIdx.x << 4;
  int nt = blockIdx.y;
  int arow = m0 + (lane & 15);
  if (arow >= M) arow = M - 1;
  const __bf16* Ar = A + arow * lda;
  v8f acc = (v8f){0, 0, 0, 0, 0, 0, 0, 0};
  for (int k0 = 0; k0 < Klen; k0 += 32) {
    v16bf fa = frag_a_rowptr(Ar, lane, k0);
    v16bf fb = frag_b_packed(Wp, lane, k0 >> 5, nt, ntilesN);
    acc = wmma_bf16(fa, fb, acc);
  }
  int col = (nt << 4) + (lane & 15);
  float bv = (bias && col < Nlogic) ? bias[col] : 0.0f;
#pragma unroll
  for (int r = 0; r < 8; ++r) {
    int row = m0 + ((lane >> 4) << 3) + r;
    if (row >= M) continue;
    float v = acc[r] + bv;
    if (relu) v = fmaxf(v, 0.0f);
    if (outB) outB[row * ldob + col] = (__bf16)v;
    if (outF && col < Nlogic) outF[row * ldof + col] = v;
  }
}

// ---------------------------------------------------------------------------
// Fused EdgeConv per 16-edge block:
//   stage A = [x_dst | x_src - x_dst] in LDS (async global->LDS when d%32==0),
//   m1 = relu(A @ W1 + b1)  (WMMA -> LDS bf16),
//   m2 = m1 @ W2 + b2       (WMMA),
//   atomic segment-max of m2 into agg[dst].
// ---------------------------------------------------------------------------
#define EC_MAXK 512
#define EC_MAXH 256
__global__ __launch_bounds__(256) void k_edgeconv(
    const __bf16* __restrict__ X, int ldx, int d,
    const __bf16* __restrict__ W1p, int Kpad, int Hlogic, int Hpad,
    const float* __restrict__ b1,
    const __bf16* __restrict__ W2p, int Ologic, const float* __restrict__ b2,
    const int* __restrict__ src, const int* __restrict__ dst, int E,
    float* __restrict__ agg, int ldagg) {
  __shared__ __align__(16) __bf16 sA[16 * EC_MAXK];
  __shared__ __align__(16) __bf16 sM[16 * EC_MAXH];
#if HAS_ASYNC_LDS
  __shared__ __align__(16) __bf16 sXs[16 * EC_MAXH];
#endif
  __shared__ int sTOff[16], sSOff[16], sDstOff[16];
  int tid = threadIdx.x;
  int e0 = blockIdx.x << 4;
  if (tid < 16) {
    int e = e0 + tid;
    if (e >= E) e = E - 1;
    sTOff[tid] = dst[e] * ldx;
    sSOff[tid] = src[e] * ldx;
    sDstOff[tid] = dst[e] * ldagg;
  }
  __syncthreads();

#if HAS_ASYNC_LDS
  if ((d & 31) == 0) {
    // async DMA rows into LDS: x_dst -> first half of sA rows, x_src -> sXs
    int csh = (d == 64) ? 3 : 5;       // 16B chunks per row = d/8
    int cpr = d >> 3;
    for (int q = tid; q < (cpr << 4); q += 256) {
      int r = q >> csh, j = (q & (cpr - 1)) << 3;
      async_copy16(X + sTOff[r] + j, sA + r * Kpad + j);
      async_copy16(X + sSOff[r] + j, sXs + r * d + j);
    }
    __builtin_amdgcn_s_wait_asynccnt(0);
    __syncthreads();
    // second half of each A row: xs - xi   (Kpad == 2d here)
    int dsh = (d == 64) ? 6 : 8;
    for (int q = tid; q < (d << 4); q += 256) {
      int r = q >> dsh, c = q & (d - 1);
      float xi = (float)sA[r * Kpad + c];
      float xs = (float)sXs[r * d + c];
      sA[r * Kpad + d + c] = (__bf16)(xs - xi);
    }
  } else
#endif
  {
    for (int i = tid; i < 16 * Kpad; i += 256) {
      int r = i >> 5, c = i & (Kpad - 1);      // scalar path only for Kpad==32
      float v = 0.0f;
      if (e0 + r < E && c < 2 * d) {
        if (c < d) v = (float)X[sTOff[r] + c];
        else       v = (float)X[sSOff[r] + (c - d)] - (float)X[sTOff[r] + (c - d)];
      }
      sA[r * Kpad + c] = (__bf16)v;
    }
  }
  __syncthreads();

  int wave = tid >> 5, lane = tid & 31;
  int row16 = lane & 15;
  bool full = (e0 + 16 <= E);
  int base = (lane >> 4) << 3;
  // GEMM1: 16 x Hpad
  int ntH = Hpad >> 4;
  for (int nt = wave; nt < ntH; nt += 8) {
    v8f acc = (v8f){0, 0, 0, 0, 0, 0, 0, 0};
    for (int k0 = 0; k0 < Kpad; k0 += 32) {
      v16bf fa = frag_a_rowptr(sA + row16 * Kpad, lane, k0);
      v16bf fb = frag_b_packed(W1p, lane, k0 >> 5, nt, ntH);
      acc = wmma_bf16(fa, fb, acc);
    }
    int col = (nt << 4) + (lane & 15);
    float bv = (col < Hlogic) ? b1[col] : 0.0f;
#pragma unroll
    for (int r = 0; r < 8; ++r)
      sM[(base + r) * Hpad + col] = (__bf16)fmaxf(acc[r] + bv, 0.0f);
  }
  __syncthreads();
  // GEMM2: 16 x Ologic, scatter with atomic max
  int ntO = Ologic >> 4;
  for (int nt = wave; nt < ntO; nt += 8) {
    v8f acc = (v8f){0, 0, 0, 0, 0, 0, 0, 0};
    for (int k0 = 0; k0 < Hpad; k0 += 32) {
      v16bf fa = frag_a_rowptr(sM + row16 * Hpad, lane, k0);
      v16bf fb = frag_b_packed(W2p, lane, k0 >> 5, nt, ntO);
      acc = wmma_bf16(fa, fb, acc);
    }
    int col = (nt << 4) + (lane & 15);
    float bv = b2[col];
    if (full) {
#pragma unroll
      for (int r = 0; r < 8; ++r)
        atomic_max_f32(&agg[sDstOff[base + r] + col], acc[r] + bv);
    } else {
#pragma unroll
      for (int r = 0; r < 8; ++r)
        if (e0 + base + r < E)
          atomic_max_f32(&agg[sDstOff[base + r] + col], acc[r] + bv);
    }
  }
}

// ---------------------------------------------------------------------------
// BN statistics (population mean/var over rows of relu(X)), one block / channel
// ---------------------------------------------------------------------------
__global__ __launch_bounds__(256) void k_stats(const float* __restrict__ X, int M, int C, int ldx,
                                               float* __restrict__ mean, float* __restrict__ var) {
  int c = blockIdx.x;
  float s1 = 0.0f, s2 = 0.0f;
  for (int r = threadIdx.x; r < M; r += 256) {
    float v = fmaxf(X[r * ldx + c], 0.0f);
    s1 += v; s2 += v * v;
  }
  __shared__ float a1[256], a2[256];
  a1[threadIdx.x] = s1; a2[threadIdx.x] = s2;
  __syncthreads();
  for (int off = 128; off > 0; off >>= 1) {
    if (threadIdx.x < off) { a1[threadIdx.x] += a1[threadIdx.x + off]; a2[threadIdx.x] += a2[threadIdx.x + off]; }
    __syncthreads();
  }
  if (threadIdx.x == 0) {
    float m = a1[0] / (float)M;
    mean[c] = m;
    var[c] = a2[0] / (float)M - m * m;
  }
}

// y = bn(relu(X)); grid(1, M), CW threads; zero-pads bf16 cols [C, CW)
__global__ void k_bn(const float* __restrict__ X, int C, int ldx,
                     const float* __restrict__ g, const float* __restrict__ b,
                     const float* __restrict__ mean, const float* __restrict__ var,
                     __bf16* outB, int ldob, float* outF, int ldof) {
  int r = blockIdx.y, c = threadIdx.x;
  if (c < C) {
    float v = fmaxf(X[r * ldx + c], 0.0f);
    float y = (v - mean[c]) * rsqrtf(var[c] + 1e-5f) * g[c] + b[c];
    if (outB) outB[r * ldob + c] = (__bf16)y;
    if (outF) outF[r * ldof + c] = y;
  } else if (outB) {
    outB[r * ldob + c] = (__bf16)0.0f;
  }
}

// pooling: grid(1, N), 256 threads
__global__ void k_pool(const float* __restrict__ xe4, const float* __restrict__ x4,
                       const int* __restrict__ batch, float* __restrict__ pooled) {
  int n = blockIdx.y, c = threadIdx.x;
  float v = (c < 208) ? xe4[n * 208 + c] : x4[n * 48 + (c - 208)];
  atomicAdd(&pooled[batch[n] * 256 + c], v);
}

// ---------------------------------------------------------------------------
// Host driver
// ---------------------------------------------------------------------------
extern "C" void kernel_launch(void* const* d_in, const int* in_sizes, int n_in,
                              void* d_out, int out_size, void* d_ws, size_t ws_size,
                              hipStream_t stream) {
  const int N  = in_sizes[0] / 8;
  const int E  = in_sizes[1] / 2;
  const int Ee = in_sizes[4] / 2;

  const int*   x_nodes  = (const int*)d_in[0];
  const int*   ei       = (const int*)d_in[1];
  const int*   etype    = (const int*)d_in[2];
  const float* pos      = (const float*)d_in[3];
  const int*   eie      = (const int*)d_in[4];
  const int*   batch    = (const int*)d_in[5];
  const float* emb1 = (const float*)d_in[6];
  const float* emb2 = (const float*)d_in[7];
  const float* emb3 = (const float*)d_in[8];
  const float* emb4 = (const float*)d_in[9];
  const float* edge_emb = (const float*)d_in[10];
  const float* G[4][7];
  for (int L = 0; L < 4; ++L)
    for (int j = 0; j < 7; ++j) G[L][j] = (const float*)d_in[11 + 7 * L + j];
  const float* BNG[4][2];
  for (int L = 0; L < 4; ++L)
    for (int j = 0; j < 2; ++j) BNG[L][j] = (const float*)d_in[39 + 2 * L + j];
  const float* EC[4][4];
  for (int L = 0; L < 4; ++L)
    for (int j = 0; j < 4; ++j) EC[L][j] = (const float*)d_in[47 + 4 * L + j];
  const float* BNE[4][2];
  for (int L = 0; L < 4; ++L)
    for (int j = 0; j < 2; ++j) BNE[L][j] = (const float*)d_in[63 + 2 * L + j];
  const float* bn6_g = (const float*)d_in[71];
  const float* bn6_b = (const float*)d_in[72];

  const int* src  = ei;
  const int* dstp = ei + E;
  const int* se   = eie;
  const int* de   = eie + Ee;

  size_t off = 0;
  auto alloc = [&](size_t bytes) -> void* {
    off = (off + 255) & ~(size_t)255;
    void* p = (char*)d_ws + off;
    off += bytes;
    return p;
  };
  __bf16* wg1a = (__bf16*)alloc(96 * 96 * 2);   __bf16* wg1b = (__bf16*)alloc(96 * 32 * 2);
  __bf16* wg2a = (__bf16*)alloc(32 * 128 * 2);  __bf16* wg2b = (__bf16*)alloc(128 * 32 * 2);
  __bf16* wg3a = (__bf16*)alloc(32 * 128 * 2);  __bf16* wg3b = (__bf16*)alloc(128 * 32 * 2);
  __bf16* wg4a = (__bf16*)alloc(32 * 32 * 2);   __bf16* wg4b = (__bf16*)alloc(32 * 48 * 2);
  __bf16* wlw[4];
  int kip[4] = {96, 32, 32, 32};
  for (int L = 0; L < 4; ++L) wlw[L] = (__bf16*)alloc(32 * kip[L] * 2);
  __bf16* we1a = (__bf16*)alloc(32 * 64 * 2);   __bf16* we1b = (__bf16*)alloc(64 * 64 * 2);
  __bf16* we2a = (__bf16*)alloc(128 * 256 * 2); __bf16* we2b = (__bf16*)alloc(256 * 256 * 2);
  __bf16* we3a = (__bf16*)alloc(512 * 256 * 2); __bf16* we3b = (__bf16*)alloc(256 * 256 * 2);
  __bf16* we4a = (__bf16*)alloc(512 * 224 * 2); __bf16* we4b = (__bf16*)alloc(224 * 208 * 2);
  __bf16* xg   = (__bf16*)alloc((size_t)N * 96 * 2);
  __bf16* tbuf = (__bf16*)alloc((size_t)N * 96 * 2);
  __bf16* hb   = (__bf16*)alloc((size_t)N * 128 * 2);
  float*  pre  = (float*)alloc((size_t)N * 48 * 4);
  float*  aggg = (float*)alloc((size_t)N * 96 * 4);
  __bf16* ea   = (__bf16*)alloc((size_t)E * 32 * 2);
  __bf16* posb = (__bf16*)alloc((size_t)N * 4 * 2);
  __bf16* xeA  = (__bf16*)alloc((size_t)N * 256 * 2);
  __bf16* xeB  = (__bf16*)alloc((size_t)N * 256 * 2);
  float*  agge = (float*)alloc((size_t)N * 256 * 4);
  float*  xe4f = (float*)alloc((size_t)N * 208 * 4);
  float*  x4f  = (float*)alloc((size_t)N * 48 * 4);
  float*  pooled = (float*)alloc(256 * 256 * 4);
  float*  mean = (float*)alloc(256 * 4);
  float*  var  = (float*)alloc(256 * 4);

  auto blocks = [](long n, int b) { return (unsigned)((n + b - 1) / b); };

  struct PW { const float* s; int K, Nl, Kp, Np; __bf16* d; };
  PW pw[] = {
    {G[0][2], 65, 65, 96, 96, wg1a},  {G[0][4], 65, 32, 96, 32, wg1b},
    {G[1][2], 32, 128, 32, 128, wg2a},{G[1][4], 128, 32, 128, 32, wg2b},
    {G[2][2], 32, 128, 32, 128, wg3a},{G[2][4], 128, 32, 128, 32, wg3b},
    {G[3][2], 32, 32, 32, 32, wg4a},  {G[3][4], 32, 48, 32, 48, wg4b},
    {G[0][0], 16, 65, 32, 96, wlw[0]}, {G[1][0], 16, 32, 32, 32, wlw[1]},
    {G[2][0], 16, 32, 32, 32, wlw[2]}, {G[3][0], 16, 32, 32, 32, wlw[3]},
    {EC[0][0], 6, 64, 32, 64, we1a},  {EC[0][2], 64, 64, 64, 64, we1b},
    {EC[1][0], 128, 256, 128, 256, we2a}, {EC[1][2], 256, 256, 256, 256, we2b},
    {EC[2][0], 512, 256, 512, 256, we3a}, {EC[2][2], 256, 256, 256, 256, we3b},
    {EC[3][0], 512, 208, 512, 224, we4a}, {EC[3][2], 208, 208, 224, 208, we4b},
  };
  for (auto& w : pw)
    k_pack_wfrag<<<blocks((long)w.Kp * w.Np, 256), 256, 0, stream>>>(w.s, w.K, w.Nl, w.d, w.Kp, w.Np);

  k_nodefeat<<<dim3(1, N), 96, 0, stream>>>(x_nodes, emb1, emb2, emb3, emb4, xg);
  k_edgefeat<<<blocks((long)E * 32, 256), 256, 0, stream>>>(etype, edge_emb, ea, E);
  k_posb<<<blocks((long)N * 4, 256), 256, 0, stream>>>(pos, posb, N);

  struct GL { int i, Kp, h, Hp, o; __bf16 *W1, *W2; };
  GL gl[4] = {
    {65, 96, 65, 96, 32, wg1a, wg1b},
    {32, 32, 128, 128, 32, wg2a, wg2b},
    {32, 32, 128, 128, 32, wg3a, wg3b},
    {32, 32, 32, 32, 48, wg4a, wg4b},
  };
  int tilesM = (N + 15) / 16;
  for (int L = 0; L < 4; ++L) {
    GL& g = gl[L];
    k_fill<<<blocks((long)N * 96, 256), 256, 0, stream>>>(aggg, (long)N * 96, 0.0f);
    k_gine_msg<<<blocks(E, 16), 256, 0, stream>>>(xg, 96, ea, wlw[L], G[L][1], g.i, g.Kp / 16,
                                                  src, dstp, E, aggg, 96);
    k_combine<<<dim3(1, N), g.Kp, 0, stream>>>(xg, 96, aggg, 96, G[L][6], g.i, g.Kp, tbuf);
    dim3 g1(tilesM, g.Hp / 16);
    k_gemm<<<g1, 32, 0, stream>>>(tbuf, g.Kp, g.W1, g.Hp / 16, G[L][3], g.h, N, g.Kp,
                                  hb, g.Hp, nullptr, 0, 1);
    dim3 g2(tilesM, g.o / 16);
    k_gemm<<<g2, 32, 0, stream>>>(hb, g.Hp, g.W2, g.o / 16, G[L][5], g.o, N, g.Hp,
                                  nullptr, 0, pre, g.o, 0);
    k_stats<<<g.o, 256, 0, stream>>>(pre, N, g.o, g.o, mean, var);
    k_bn<<<dim3(1, N), 96, 0, stream>>>(pre, g.o, g.o, BNG[L][0], BNG[L][1], mean, var,
                                        xg, 96, (L == 3) ? x4f : nullptr, 48);
  }

  struct EL { const __bf16* X; int ldx, d, Kp, H, Hp, O; __bf16 *W1, *W2; __bf16* outX; float* outF; };
  EL el[4] = {
    {posb, 4, 3, 32, 64, 64, 64, we1a, we1b, xeA, nullptr},
    {xeA, 256, 64, 128, 256, 256, 256, we2a, we2b, xeB, nullptr},
    {xeB, 256, 256, 512, 256, 256, 256, we3a, we3b, xeA, nullptr},
    {xeA, 256, 256, 512, 208, 224, 208, we4a, we4b, nullptr, xe4f},
  };
  for (int L = 0; L < 4; ++L) {
    EL& e = el[L];
    k_fill<<<blocks((long)N * 256, 256), 256, 0, stream>>>(agge, (long)N * 256, -HUGE_VALF);
    k_edgeconv<<<blocks(Ee, 16), 256, 0, stream>>>(
        e.X, e.ldx, e.d, e.W1, e.Kp, e.H, e.Hp, EC[L][1], e.W2, e.O, EC[L][3],
        se, de, Ee, agge, 256);
    k_stats<<<e.O, 256, 0, stream>>>(agge, N, e.O, 256, mean, var);
    int CW = e.outX ? 256 : e.O;
    k_bn<<<dim3(1, N), CW, 0, stream>>>(agge, e.O, 256, BNE[L][0], BNE[L][1], mean, var,
                                        e.outX, 256, e.outF, 208);
  }

  k_fill<<<blocks(256 * 256, 256), 256, 0, stream>>>(pooled, 256 * 256, 0.0f);
  k_pool<<<dim3(1, N), 256, 0, stream>>>(xe4f, x4f, batch, pooled);
  k_stats<<<256, 256, 0, stream>>>(pooled, 256, 256, 256, mean, var);
  k_bn<<<dim3(1, 256), 256, 0, stream>>>(pooled, 256, 256, bn6_g, bn6_b, mean, var,
                                         nullptr, 0, (float*)d_out, 256);
  (void)n_in; (void)in_sizes; (void)ws_size; (void)out_size;
}
